// QuantLinear_24721831756581
// MI455X (gfx1250) — matrix operand order repaired and checked by reading, not compiled, not run on hardware
//
#include <hip/hip_runtime.h>

// ---------------------------------------------------------------------------
// QuantLinear (int4 groupwise dequant + GEMM) for gfx1250 (MI455X, wave32).
//   out[M,N] = x[M,K] * W[N,K]^T + bias,  W = dequant_int4(packed,scales,offsets)
//   M = 8192, N = 4096, K = 4096, GROUP = 128
// bf16 WMMA (v_wmma_f32_16x16x32_bf16), 128x128 block tile, 8 waves,
// 64x32 wave tile (4x2 WMMA tiles), K-step 32.
// Software pipeline: global loads for step kt+1 are issued into registers
// BEFORE the WMMAs of step kt; the workgroup barrier is a hand-rolled
// "s_wait_dscnt 0 + s_barrier" so it does NOT drain the in-flight global
// loads (unlike __syncthreads, which waits loadcnt to 0). One barrier per
// K-step is sufficient: dscnt-0 at the barrier covers both the previous
// step's fragment ds_loads and this step's staging ds_stores (DS is
// in-order per wave), and each buffer's writer/reader pair is separated
// by at least one barrier.
// ---------------------------------------------------------------------------

typedef __attribute__((ext_vector_type(16))) __bf16 v16bf;
typedef __attribute__((ext_vector_type(2)))  __bf16 v2bf;
typedef __attribute__((ext_vector_type(8)))  float  v8f;

#define M_DIM 8192
#define N_DIM 4096
#define K_DIM 4096
#define TILE_M 128
#define TILE_N 128
#define TILE_K 32
#define NKT (K_DIM / TILE_K)   // 128 K-steps

union FragB16 {
    v16bf v;
    uint4 q[2];
};

// Workgroup barrier that only orders LDS traffic: waits DScnt==0 then
// signals/waits the workgroup barrier. Crucially does NOT wait LOADcnt,
// so global loads issued for the next K-step keep flying across it.
__device__ __forceinline__ void wg_barrier_lds() {
    asm volatile("s_wait_dscnt 0x0\n\t"
                 "s_barrier_signal -1\n\t"
                 "s_barrier_wait -1" ::: "memory");
}

// Pack two fp32 into one dword of two bf16 (lo -> [15:0], hi -> [31:16]).
__device__ __forceinline__ unsigned int pack2bf(float lo, float hi) {
#if __has_builtin(__builtin_amdgcn_cvt_pk_bf16_f32)
    v2bf p = __builtin_amdgcn_cvt_pk_bf16_f32(lo, hi);
    return __builtin_bit_cast(unsigned int, p);
#else
    // round-half-up then grab high halves of both floats with one v_perm_b32
    union { float f; unsigned int u; } a, b;
    a.f = lo; b.f = hi;
    return __builtin_amdgcn_perm(b.u + 0x8000u, a.u + 0x8000u, 0x07060302u);
#endif
}

__global__ __launch_bounds__(256) void QuantLinear_wmma_kernel(
    const float* __restrict__ x,        // [M, K] fp32
    const int*   __restrict__ packed,   // [M*K/2] one byte per int32 (2 int4)
    const float* __restrict__ scales,   // [M*K/128]
    const float* __restrict__ offsets,  // [M*K/128]
    const float* __restrict__ bias,     // [N]
    float*       __restrict__ out)      // [M, N] fp32
{
    // Fragment-ready LDS: [buf][tile16][lane][16 halves]  (16 KB per array)
    __shared__ unsigned short Afrag[2][8][32][16];
    __shared__ unsigned short Bfrag[2][8][32][16];

    const int t    = threadIdx.x;
    const int lane = t & 31;
    const int wid  = t >> 5;     // 8 waves
    const int wm   = wid & 1;    // wave row   (2 x 64 rows)
    const int wn   = wid >> 1;   // wave col   (4 x 32 cols)
    const int n0   = blockIdx.x * TILE_N;
    const int m0   = blockIdx.y * TILE_M;

    // ---- loop-invariant staging coordinates (hoisted) ----
    // A: 4 chunks of float4; chunk c = t + 256j -> row r, K offset kq.
    const float*    a_gsrc[4];
    unsigned short* a_ldst[4];
#pragma unroll
    for (int j = 0; j < 4; ++j) {
        const int c  = t + 256 * j;       // 0..1023
        const int r  = c >> 3;            // row in tile (0..127)
        const int kq = (c & 7) << 2;      // K offset (0,4,..,28)
        a_gsrc[j] = x + (size_t)(m0 + r) * K_DIM + kq;
        // A 16x32 bf16 fragment layout (ISA 7.12.2):
        //   chunk = K/8: 0 -> lane row    elems 0-7 ; 1 -> lane row+16 elems 0-7
        //                2 -> lane row    elems 8-15; 3 -> lane row+16 elems 8-15
        const int chunk = kq >> 3;
        const int laned = (r & 15) + ((chunk & 1) << 4);
        const int elem  = ((chunk >> 1) << 3) + (kq & 7);
        a_ldst[j] = &Afrag[0][r >> 4][laned][elem];
    }
    // B: 2 chunks of 8 int4 weights; chunk c = t + 256j -> weight row n, K off i.
    const int*      b_gsrc[2];
    const float*    s_src[2];
    const float*    o_src[2];
    unsigned short* b_ldst[2];
#pragma unroll
    for (int j = 0; j < 2; ++j) {
        const int c = t + 256 * j;        // 0..511
        const int n = c >> 2;             // weight row in tile (0..127)
        const int i = (c & 3) << 3;       // K offset (0,8,16,24)
        const long long flat = (long long)(n0 + n) * K_DIM + i;
        b_gsrc[j] = packed + (flat >> 1);
        // group index g = (flat + kb)>>7 = (n0+n)*32 + (kb>>7): the low bits
        // (kb&127)+i <= 120 never carry, so the per-K-step part is uniform.
        s_src[j] = scales  + (size_t)(n0 + n) * (K_DIM / 128);
        o_src[j] = offsets + (size_t)(n0 + n) * (K_DIM / 128);
        // B 32x16 bf16 fragment layout: lanes 0-15 hold K 0-7 (elems 0-7) and
        // K 16-23 (elems 8-15); lanes 16-31 hold K 8-15 and K 24-31.
        const int laned = (n & 15) + (((i >> 3) & 1) << 4);
        const int elem  = (i >> 4) << 3;
        b_ldst[j] = &Bfrag[0][n >> 4][laned][elem];
    }
    const int ldsbuf_halves = 8 * 32 * 16;  // 4096 halves = 8 KB per buffer

    v8f acc[4][2];
#pragma unroll
    for (int mt = 0; mt < 4; ++mt)
#pragma unroll
        for (int nt = 0; nt < 2; ++nt) {
            v8f z = {};
            acc[mt][nt] = z;
        }

    // ---- pipeline stage 1: issue global loads into registers (no wait) ----
    float4 areg[4];
    int4   breg[2];
    float  sreg[2], oreg[2];
    auto issue_loads = [&](int kt) {
        const int kb = kt * TILE_K;
#pragma unroll
        for (int j = 0; j < 4; ++j) {
            const float* gp = a_gsrc[j] + kb;
            if (kt + 2 < NKT)                 // warm L2 two steps ahead
                __builtin_prefetch(gp + 2 * TILE_K, 0, 1);
            areg[j] = *(const float4*)gp;
        }
        const int gk = kt >> 2;               // uniform group column
#pragma unroll
        for (int j = 0; j < 2; ++j) {
            const int* pp = b_gsrc[j] + (kb >> 1);
            if (kt + 2 < NKT)
                __builtin_prefetch(pp + TILE_K, 0, 1);
            breg[j] = *(const int4*)pp;
            sreg[j] = s_src[j][gk];
            oreg[j] = o_src[j][gk];
        }
    };

    // ---- pipeline stage 2: convert/dequant pending regs, store to LDS ----
    auto store_lds = [&](int buf) {
        const int boff = buf * ldsbuf_halves;
#pragma unroll
        for (int j = 0; j < 4; ++j) {
            const unsigned int d0 = pack2bf(areg[j].x, areg[j].y);
            const unsigned int d1 = pack2bf(areg[j].z, areg[j].w);
            *(uint2*)(a_ldst[j] + boff) = make_uint2(d0, d1);
        }
#pragma unroll
        for (int j = 0; j < 2; ++j) {
            const float s  = sreg[j];
            const float of = oreg[j];
            const int   pv[4] = {breg[j].x, breg[j].y, breg[j].z, breg[j].w};
            unsigned int d[4];
#pragma unroll
            for (int b = 0; b < 4; ++b) {
                const float wlo = (float)(pv[b] & 15) * s + of;         // even idx
                const float whi = (float)((pv[b] >> 4) & 15) * s + of;  // odd idx
                d[b] = pack2bf(wlo, whi);
            }
            *(uint4*)(b_ldst[j] + boff) = make_uint4(d[0], d[1], d[2], d[3]);
        }
    };

    issue_loads(0);

    for (int kt = 0; kt < NKT; ++kt) {
        const int buf = kt & 1;
        store_lds(buf);                         // waits only on loads issued
        wg_barrier_lds();                       //   one compute-phase earlier
        if (kt + 1 < NKT) issue_loads(kt + 1);  // fire-and-forget

        FragB16 a[4], b[2];
#pragma unroll
        for (int mt = 0; mt < 4; ++mt) {
            const uint4* p = (const uint4*)&Afrag[buf][wm * 4 + mt][lane][0];
            a[mt].q[0] = p[0];
            a[mt].q[1] = p[1];
        }
#pragma unroll
        for (int nt = 0; nt < 2; ++nt) {
            const uint4* p = (const uint4*)&Bfrag[buf][wn * 2 + nt][lane][0];
            b[nt].q[0] = p[0];
            b[nt].q[1] = p[1];
        }
#pragma unroll
        for (int mt = 0; mt < 4; ++mt)
#pragma unroll
            for (int nt = 0; nt < 2; ++nt)
                acc[mt][nt] = __builtin_amdgcn_wmma_f32_16x16x32_bf16(
                    false, a[mt].v, false, b[nt].v,
                    (short)0, acc[mt][nt], false, false);
    }

    // ---- epilogue: D layout (ISA 7.12.2): lane -> N, VGPR v -> M ----
    const int laneN = lane & 15;
    const int rsel  = (lane >> 4) << 3;  // lanes 16-31 hold M = 8..15
#pragma unroll
    for (int nt = 0; nt < 2; ++nt) {
        const int col = n0 + wn * 32 + nt * 16 + laneN;
        const float bv = bias[col];
#pragma unroll
        for (int mt = 0; mt < 4; ++mt) {
#pragma unroll
            for (int v = 0; v < 8; ++v) {
                const int row = m0 + wm * 64 + mt * 16 + rsel + v;
                out[(size_t)row * N_DIM + col] = acc[mt][nt][v] + bv;
            }
        }
    }
}

extern "C" void kernel_launch(void* const* d_in, const int* in_sizes, int n_in,
                              void* d_out, int out_size, void* d_ws, size_t ws_size,
                              hipStream_t stream) {
    (void)in_sizes; (void)n_in; (void)out_size; (void)d_ws; (void)ws_size;
    const float* x       = (const float*)d_in[0];
    const int*   packed  = (const int*)d_in[1];
    const float* scales  = (const float*)d_in[2];
    const float* offsets = (const float*)d_in[3];
    const float* bias    = (const float*)d_in[4];
    float*       out     = (float*)d_out;

    dim3 grid(N_DIM / TILE_N, M_DIM / TILE_M);  // 32 x 64 workgroups
    QuantLinear_wmma_kernel<<<grid, 256, 0, stream>>>(
        x, packed, scales, offsets, bias, out);
}
